// L1RegressionActionHead_68719477547
// MI455X (gfx1250) — compile-verified, matrix operand order checked by reading, not run on hardware
//
#include <hip/hip_runtime.h>

// ---------------------------------------------------------------------------
// MI455X / gfx1250 implementation.
//  - GEMMs: V_WMMA_F32_16X16X32_F16, f16 operands staged into LDS with
//    GLOBAL_LOAD_ASYNC_TO_LDS_B128 (ASYNCcnt), double-buffered pipeline.
//  - All GEMM operands pre-converted to f16; f32 masters kept for LN/residual.
// ---------------------------------------------------------------------------

typedef __attribute__((ext_vector_type(16))) _Float16 v16h;
typedef __attribute__((ext_vector_type(8)))  _Float16 v8h;
typedef __attribute__((ext_vector_type(4)))  _Float16 v4h;
typedef __attribute__((ext_vector_type(8)))  float    v8f;
typedef __attribute__((ext_vector_type(4)))  float    v4f;

#define DIM    896
#define HEADS  8
#define HD     112
#define BSZ    64
#define TSEQ   8
#define KA     64
#define KT     512
#define KADP   65                     // KA + 1 (p row appended)
#define KTOT   (TSEQ + KADP + KT)     // 585 total keys

#define BM   128
#define BN   128
#define BKK  32
#define LDSS 40                       // padded LDS row stride (halves) -> 80B

#define GF_RELU 1
#define GF_RES  2

// low 32 bits of a generic pointer to LDS == LDS byte offset (ISA 10.2)
__device__ __forceinline__ unsigned lds_off(const void* p) {
    return (unsigned)(size_t)p;
}
__device__ __forceinline__ void async_b128(unsigned lds, const _Float16* g) {
    asm volatile("global_load_async_to_lds_b128 %0, %1, off"
                 :: "v"(lds), "v"(g) : "memory");
}

// ===========================================================================
// WMMA GEMM: C[M,N] = A[M,K](f16) @ W[N,K]^T(f16) (+bias)(+res)(relu)
// Tiles staged LDS via async copies, double buffered. A/W rows packed.
// ===========================================================================
__global__ __launch_bounds__(256)
void gemm_wmma(const _Float16* __restrict__ A, int lda,
               const _Float16* __restrict__ W, int ldw,
               const float* __restrict__ bias,
               const float* __restrict__ res,
               float* __restrict__ out32, _Float16* __restrict__ out16,
               int ldc, int M, int N, int K, int flags)
{
    __shared__ _Float16 As[2][BM * LDSS];
    __shared__ _Float16 Bs[2][BN * LDSS];

    const int tid  = threadIdx.x;
    const int lane = tid & 31;
    const int wid  = tid >> 5;     // 8 waves
    const int wm   = wid & 3;      // 4 wave-rows  (32 rows each)
    const int wn   = wid >> 2;     // 2 wave-cols  (64 cols each)
    const int tm0  = blockIdx.x * BM;
    const int tn0  = blockIdx.y * BN;

    v8f acc[2][4];
#pragma unroll
    for (int i = 0; i < 2; ++i)
#pragma unroll
        for (int j = 0; j < 4; ++j) acc[i][j] = v8f{};

    // global->LDS mapping: thread covers (row = tid/2, 16 consecutive halves)
    const int row_l = tid >> 1;
    const int kh    = (tid & 1) * 16;

    // clamp OOB rows to a valid row: keeps EXEC uniform; junk rows never stored
    int gr = tm0 + row_l; if (gr >= M) gr = M - 1;
    int gn = tn0 + row_l; if (gn >= N) gn = N - 1;
    const _Float16* arow = A + (size_t)gr * lda + kh;
    const _Float16* wrow = W + (size_t)gn * ldw + kh;

    const unsigned aslot[2] = { lds_off(&As[0][row_l * LDSS + kh]),
                                lds_off(&As[1][row_l * LDSS + kh]) };
    const unsigned bslot[2] = { lds_off(&Bs[0][row_l * LDSS + kh]),
                                lds_off(&Bs[1][row_l * LDSS + kh]) };

    auto issue = [&](int buf, int k0) {
        const _Float16* ga = arow + k0;
        const _Float16* gw = wrow + k0;
        unsigned la = buf ? aslot[1] : aslot[0];
        unsigned lb = buf ? bslot[1] : bslot[0];
        async_b128(la,      ga);
        async_b128(la + 16, ga + 8);
        async_b128(lb,      gw);
        async_b128(lb + 16, gw + 8);
    };

    // fragment gather coords (ISA 7.12.2 16-bit A/B layout)
    const int fr = lane & 15;
    const int kb = (lane >> 4) * 8;
    const int steps = K / BKK;     // 28

    issue(0, 0);
    for (int s = 0; s < steps; ++s) {
        const int buf = s & 1;
        if (s + 1 < steps) {
            issue(buf ^ 1, (s + 1) * BKK);
            asm volatile("s_wait_asynccnt 0x4" ::: "memory");  // retire buf s
        } else {
            asm volatile("s_wait_asynccnt 0x0" ::: "memory");
        }
        __syncthreads();           // tile visible to all waves

        const _Float16* asb = &As[buf][0];
        const _Float16* bsb = &Bs[buf][0];
        v16h af[2];
#pragma unroll
        for (int fm = 0; fm < 2; ++fm) {
            const _Float16* p = asb + (wm * 32 + fm * 16 + fr) * LDSS + kb;
            v8h c0 = *(const v8h*)p;
            v8h c1 = *(const v8h*)(p + 16);
            af[fm] = __builtin_shufflevector(c0, c1, 0,1,2,3,4,5,6,7,8,9,10,11,12,13,14,15);
        }
        v16h bf[4];
#pragma unroll
        for (int fn = 0; fn < 4; ++fn) {
            const _Float16* p = bsb + (wn * 64 + fn * 16 + fr) * LDSS + kb;
            v8h c0 = *(const v8h*)p;
            v8h c1 = *(const v8h*)(p + 16);
            bf[fn] = __builtin_shufflevector(c0, c1, 0,1,2,3,4,5,6,7,8,9,10,11,12,13,14,15);
        }
#pragma unroll
        for (int fm = 0; fm < 2; ++fm)
#pragma unroll
            for (int fn = 0; fn < 4; ++fn)
                acc[fm][fn] = __builtin_amdgcn_wmma_f32_16x16x32_f16(
                    false, af[fm], false, bf[fn], (short)0, acc[fm][fn], false, false);
        __syncthreads();           // all reads of buf done before re-fill
    }

    // epilogue: D layout -> (m = vgpr + 8*(lane>=16), n = lane&15)
    const int cn   = lane & 15;
    const int r_hi = (lane >> 4) * 8;
#pragma unroll
    for (int fm = 0; fm < 2; ++fm)
#pragma unroll
        for (int fn = 0; fn < 4; ++fn) {
            int col = tn0 + wn * 64 + fn * 16 + cn;
            if (col >= N) continue;
            float bv = bias ? bias[col] : 0.f;
#pragma unroll
            for (int r = 0; r < 8; ++r) {
                int row = tm0 + wm * 32 + fm * 16 + r_hi + r;
                if (row >= M) continue;
                float v = acc[fm][fn][r] + bv;
                if (flags & GF_RES)  v += res[(size_t)row * ldc + col];
                if (flags & GF_RELU) v = fmaxf(v, 0.f);
                if (out32) out32[(size_t)row * ldc + col] = v;
                if (out16) out16[(size_t)row * ldc + col] = (_Float16)v;
            }
        }
}

// ===========================================================================
// f32 -> f16 conversion (vec4)
// ===========================================================================
__global__ void cvt4(const float* __restrict__ src, _Float16* __restrict__ dst, int n4)
{
    int i = blockIdx.x * blockDim.x + threadIdx.x;
    if (i >= n4) return;
    v4f f = ((const v4f*)src)[i];
    ((v4h*)dst)[i] = v4h{(_Float16)f.x, (_Float16)f.y, (_Float16)f.z, (_Float16)f.w};
}

// gather h_t[:, blk+1] slice into packed (B*512, 896) f16 panel
__global__ void cvt_hts(const float* __restrict__ h_t, _Float16* __restrict__ dst, int blk)
{
    int i = blockIdx.x * blockDim.x + threadIdx.x;          // vec4 index
    int total4 = BSZ * KT * DIM / 4;
    if (i >= total4) return;
    int row = i / (DIM / 4), c4 = i - row * (DIM / 4);
    int b = row >> 9, j = row & (KT - 1);
    const v4f* s = (const v4f*)(h_t + ((size_t)(b * 3 + blk + 1) * KT + j) * DIM) + c4;
    v4f f = *s;
    ((v4h*)dst)[i] = v4h{(_Float16)f.x, (_Float16)f.y, (_Float16)f.z, (_Float16)f.w};
}

// ===========================================================================
// LayerNorm: one wave per row; optional f32 and f16 outputs
// ===========================================================================
__global__ __launch_bounds__(256)
void ln_kernel(const float* __restrict__ x, const float* __restrict__ w,
               const float* __restrict__ b, float* __restrict__ y32,
               _Float16* __restrict__ y16, int M, int C)
{
    int row  = blockIdx.x * 8 + (threadIdx.x >> 5);
    int lane = threadIdx.x & 31;
    if (row >= M) return;
    const float* xr = x + (size_t)row * C;
    float s = 0.f, s2 = 0.f;
    for (int c = lane; c < C; c += 32) { float v = xr[c]; s += v; s2 += v * v; }
    for (int off = 16; off; off >>= 1) {
        s  += __shfl_xor(s,  off, 32);
        s2 += __shfl_xor(s2, off, 32);
    }
    float mean = s / C;
    float var  = s2 / C - mean * mean;
    float inv  = rsqrtf(var + 1e-5f);
    for (int c = lane; c < C; c += 32) {
        float o = (xr[c] - mean) * inv * w[c] + b[c];
        if (y32) y32[(size_t)row * C + c] = o;
        if (y16) y16[(size_t)row * C + c] = (_Float16)o;
    }
}

// ===========================================================================
// RoPE (reference: concat cos table + interleaved rotate_half)
// ===========================================================================
template <typename T>
__global__ void rope_kernel(T* __restrict__ data, int Mrows, int L)
{
    int idx   = blockIdx.x * blockDim.x + threadIdx.x;
    int total = Mrows * HEADS * (HD / 2);       // rows * 448 pairs
    if (idx >= total) return;
    int row  = idx / 448;
    int rem  = idx - row * 448;
    int head = rem / 56;
    int i    = rem - head * 56;
    int t    = row % L;

    T* p = data + (size_t)row * DIM + head * HD + 2 * i;
    float x0 = (float)p[0], x1 = (float)p[1];

    const float LN1E4_O = 9.210340371976184f / 112.f;   // ln(10000)/HD
    int j0 = (2 * i) % 56, j1 = (2 * i + 1) % 56;
    float f0 = __expf(-(float)(2 * j0) * LN1E4_O);
    float f1 = __expf(-(float)(2 * j1) * LN1E4_O);
    float a0 = (float)t * f0, a1 = (float)t * f1;
    float n0 = x0 * cosf(a0) - x1 * sinf(a0);
    float n1 = x1 * cosf(a1) + x0 * sinf(a1);
    p[0] = (T)n0;
    p[1] = (T)n1;
}

// ===========================================================================
// Pack h_adapter = concat([h_a[:, blk+1], p], axis=1) -> (B*65, 896) f16
// ===========================================================================
__global__ void pack_adapter(const float* __restrict__ h_a,
                             const float* __restrict__ p, int blk,
                             _Float16* __restrict__ out)
{
    int idx = blockIdx.x * blockDim.x + threadIdx.x;
    int total = BSZ * KADP * DIM;
    if (idx >= total) return;
    int row = idx / DIM, c = idx - row * DIM;
    int b = row / KADP, j = row - b * KADP;
    float v = (j < KA)
        ? h_a[(((size_t)b * 3 + blk + 1) * KA + j) * DIM + c]
        : p[(size_t)b * DIM + c];
    out[idx] = (_Float16)v;
}

// ===========================================================================
// Attention: one workgroup per (b, h). 585 keys = [8 tok | 65 ad | 512 ta].
// ===========================================================================
__global__ __launch_bounds__(256)
void attn_kernel(const float* __restrict__ q,
                 const float* __restrict__ ktok, const float* __restrict__ vtok,
                 const _Float16* __restrict__ kad, const _Float16* __restrict__ vad,
                 const _Float16* __restrict__ kta, const _Float16* __restrict__ vta,
                 const float* __restrict__ gscale, _Float16* __restrict__ out)
{
    const int bh = blockIdx.x;
    const int b  = bh >> 3;
    const int h  = bh & 7;
    const int tid = threadIdx.x;

    __shared__ float qs[TSEQ][HD];
    __shared__ float sc[TSEQ][592];   // 585 padded

    const float ratio = tanhf(gscale[0]);
    const float scl   = 0.09449111825230680f;   // 1/sqrt(112)

    for (int i = tid; i < TSEQ * HD; i += 256) {
        int t = i / HD, d = i - t * HD;
        qs[t][d] = q[((size_t)(b * TSEQ + t)) * DIM + h * HD + d];
    }
    __syncthreads();

    for (int i = tid; i < TSEQ * KTOT; i += 256) {
        int t = i / KTOT, kk = i - t * KTOT;
        const float* qp = qs[t];
        float s = 0.f;
        if (kk < TSEQ) {
            const float* kp = ktok + ((size_t)(b * TSEQ + kk)) * DIM + h * HD;
            for (int d = 0; d < HD; ++d) s += qp[d] * kp[d];
            s *= scl;
        } else if (kk < TSEQ + KADP) {
            const _Float16* kp = kad + ((size_t)(b * KADP + kk - TSEQ)) * DIM + h * HD;
            for (int d = 0; d < HD; ++d) s += qp[d] * (float)kp[d];
            s *= scl;
        } else {
            const _Float16* kp = kta + ((size_t)(b * KT + kk - TSEQ - KADP)) * DIM + h * HD;
            for (int d = 0; d < HD; ++d) s += qp[d] * (float)kp[d];
            s *= scl * ratio;
        }
        sc[t][kk] = s;
    }
    __syncthreads();

    {   // softmax: wave t handles query row t (8 waves exactly)
        int t = tid >> 5, lane = tid & 31;
        float m = -1e30f;
        for (int kk = lane; kk < KTOT; kk += 32) m = fmaxf(m, sc[t][kk]);
        for (int off = 16; off; off >>= 1) m = fmaxf(m, __shfl_xor(m, off, 32));
        float sum = 0.f;
        for (int kk = lane; kk < KTOT; kk += 32) {
            float e = __expf(sc[t][kk] - m);
            sc[t][kk] = e;
            sum += e;
        }
        for (int off = 16; off; off >>= 1) sum += __shfl_xor(sum, off, 32);
        float rinv = 1.f / sum;
        for (int kk = lane; kk < KTOT; kk += 32) sc[t][kk] *= rinv;
    }
    __syncthreads();

    for (int i = tid; i < TSEQ * HD; i += 256) {
        int t = i / HD, d = i - t * HD;
        const float* w = sc[t];
        float s = 0.f;
        for (int j = 0; j < TSEQ; ++j)
            s += w[j] * vtok[((size_t)(b * TSEQ + j)) * DIM + h * HD + d];
        for (int j = 0; j < KADP; ++j)
            s += w[TSEQ + j] * (float)vad[((size_t)(b * KADP + j)) * DIM + h * HD + d];
        for (int j = 0; j < KT; ++j)
            s += w[TSEQ + KADP + j] * (float)vta[((size_t)(b * KT + j)) * DIM + h * HD + d];
        out[((size_t)(b * TSEQ + t)) * DIM + h * HD + d] = (_Float16)s;
    }
}

// ===========================================================================
// Final head: (512, 896) @ (7, 896)^T + b2 -> (512, 7). One wave per output.
// ===========================================================================
__global__ __launch_bounds__(256)
void final_proj(const float* __restrict__ x, const float* __restrict__ W2,
                const float* __restrict__ b2, float* __restrict__ out)
{
    int gw   = blockIdx.x * 8 + (threadIdx.x >> 5);
    int lane = threadIdx.x & 31;
    if (gw >= 512 * 7) return;
    int row = gw / 7, o = gw - row * 7;
    const float* xr = x  + (size_t)row * DIM;
    const float* wr = W2 + (size_t)o * DIM;
    float s = 0.f;
    for (int c = lane; c < DIM; c += 32) s += xr[c] * wr[c];
    for (int off = 16; off; off >>= 1) s += __shfl_xor(s, off, 32);
    if (lane == 0) out[(size_t)row * 7 + o] = s + b2[o];
}

// ===========================================================================
// Host orchestration
// ===========================================================================
extern "C" void kernel_launch(void* const* d_in, const int* in_sizes, int n_in,
                              void* d_out, int out_size, void* d_ws, size_t ws_size,
                              hipStream_t stream)
{
    (void)in_sizes; (void)n_in; (void)out_size; (void)ws_size;

    // setup_inputs() recursive insertion order (see reference file)
    const float* x_in  = (const float*)d_in[0];
    const float* h_a   = (const float*)d_in[1];
    const float* h_t   = (const float*)d_in[2];
    const float* p_in  = (const float*)d_in[3];
    const float* ln1w  = (const float*)d_in[4];
    const float* ln1b  = (const float*)d_in[5];
    const float* W1    = (const float*)d_in[6];
    const float* b1    = (const float*)d_in[7];
    const float* ln2w  = (const float*)d_in[8];
    const float* ln2b  = (const float*)d_in[9];
    const float* W2    = (const float*)d_in[10];
    const float* b2    = (const float*)d_in[11];
    const float* Wq    = (const float*)d_in[12];
    const float* p_bq  = (const float*)d_in[13];
    const float* Wks   = (const float*)d_in[14];
    const float* p_bks = (const float*)d_in[15];
    const float* Wvs   = (const float*)d_in[16];
    const float* p_bvs = (const float*)d_in[17];
    const float* Wka   = (const float*)d_in[18];
    const float* p_bka = (const float*)d_in[19];
    const float* Wva   = (const float*)d_in[20];
    const float* p_bva = (const float*)d_in[21];
    const float* Wkt   = (const float*)d_in[22];
    const float* p_bkt = (const float*)d_in[23];
    const float* Wvt   = (const float*)d_in[24];
    const float* p_bvt = (const float*)d_in[25];
    const float* Wo    = (const float*)d_in[26];
    const float* p_bo  = (const float*)d_in[27];
    const float* Wf    = (const float*)d_in[28];
    const float* p_bf  = (const float*)d_in[29];
    const float* lnw   = (const float*)d_in[30];
    const float* lnb   = (const float*)d_in[31];
    const float* p_g   = (const float*)d_in[32];

    const int NTOK = BSZ * TSEQ;     // 512
    const int NAD  = BSZ * KADP;     // 4160
    const int NTA  = BSZ * KT;       // 32768
    const int WSZ  = DIM * DIM;      // 802816

    size_t off = 0;
    auto alloc = [&](size_t bytes) -> void* {
        void* r = (char*)d_ws + off;
        off += (bytes + 255) & ~(size_t)255;
        return r;
    };
    _Float16* xc16    = (_Float16*)alloc((size_t)NTOK * DIM * 2);
    float*    xc32    = (float*)alloc((size_t)NTOK * DIM * 4);
    float*    x_cur   = (float*)alloc((size_t)NTOK * DIM * 4);
    _Float16* x_cur16 = (_Float16*)alloc((size_t)NTOK * DIM * 2);
    float*    qb      = (float*)alloc((size_t)NTOK * DIM * 4);
    float*    ktokb   = (float*)alloc((size_t)NTOK * DIM * 4);
    float*    vtokb   = (float*)alloc((size_t)NTOK * DIM * 4);
    float*    hb      = (float*)alloc((size_t)NTOK * DIM * 4);
    _Float16* hln16   = (_Float16*)alloc((size_t)NTOK * DIM * 2);
    _Float16* attn16  = (_Float16*)alloc((size_t)NTOK * DIM * 2);
    _Float16* adp16   = (_Float16*)alloc((size_t)NAD * DIM * 2);
    _Float16* kadb    = (_Float16*)alloc((size_t)NAD * DIM * 2);
    _Float16* vadb    = (_Float16*)alloc((size_t)NAD * DIM * 2);
    _Float16* hts16   = (_Float16*)alloc((size_t)NTA * DIM * 2);
    _Float16* ktab    = (_Float16*)alloc((size_t)NTA * DIM * 2);
    _Float16* vtab    = (_Float16*)alloc((size_t)NTA * DIM * 2);
    _Float16* w1h     = (_Float16*)alloc((size_t)WSZ * 2);
    _Float16* wqh     = (_Float16*)alloc((size_t)2 * WSZ * 2);
    _Float16* wksh    = (_Float16*)alloc((size_t)2 * WSZ * 2);
    _Float16* wvsh    = (_Float16*)alloc((size_t)2 * WSZ * 2);
    _Float16* wkah    = (_Float16*)alloc((size_t)2 * WSZ * 2);
    _Float16* wvah    = (_Float16*)alloc((size_t)2 * WSZ * 2);
    _Float16* wkth    = (_Float16*)alloc((size_t)2 * WSZ * 2);
    _Float16* wvth    = (_Float16*)alloc((size_t)2 * WSZ * 2);
    _Float16* woh     = (_Float16*)alloc((size_t)2 * WSZ * 2);
    _Float16* wfh     = (_Float16*)alloc((size_t)2 * WSZ * 2);

    const dim3 B256(256);
    const dim3 Gtok(4, 7);      // M=512
    const dim3 Gad(33, 7);      // M=4160
    const dim3 Gta(256, 7);     // M=32768
    const int  CB1 = (WSZ / 4 + 255) / 256;
    const int  CB2 = (2 * WSZ / 4 + 255) / 256;

    // weight conversions (f32 -> f16)
    cvt4<<<CB1, B256, 0, stream>>>(W1,  w1h,  WSZ / 4);
    cvt4<<<CB2, B256, 0, stream>>>(Wq,  wqh,  2 * WSZ / 4);
    cvt4<<<CB2, B256, 0, stream>>>(Wks, wksh, 2 * WSZ / 4);
    cvt4<<<CB2, B256, 0, stream>>>(Wvs, wvsh, 2 * WSZ / 4);
    cvt4<<<CB2, B256, 0, stream>>>(Wka, wkah, 2 * WSZ / 4);
    cvt4<<<CB2, B256, 0, stream>>>(Wva, wvah, 2 * WSZ / 4);
    cvt4<<<CB2, B256, 0, stream>>>(Wkt, wkth, 2 * WSZ / 4);
    cvt4<<<CB2, B256, 0, stream>>>(Wvt, wvth, 2 * WSZ / 4);
    cvt4<<<CB2, B256, 0, stream>>>(Wo,  woh,  2 * WSZ / 4);
    cvt4<<<CB2, B256, 0, stream>>>(Wf,  wfh,  2 * WSZ / 4);

    // x = relu(W1 @ LN1(x) + b1)
    ln_kernel<<<64, B256, 0, stream>>>(x_in, ln1w, ln1b, nullptr, xc16, NTOK, DIM);
    gemm_wmma<<<Gtok, B256, 0, stream>>>(xc16, DIM, w1h, DIM, b1, nullptr,
                                         x_cur, x_cur16, DIM, NTOK, DIM, DIM, GF_RELU);

    for (int blk = 0; blk < 2; ++blk) {
        const size_t wo = (size_t)blk * WSZ;
        const size_t vo = (size_t)blk * DIM;

        pack_adapter<<<(NAD * DIM + 255) / 256, B256, 0, stream>>>(h_a, p_in, blk, adp16);
        cvt_hts<<<(NTA * DIM / 4 + 255) / 256, B256, 0, stream>>>(h_t, hts16, blk);

        gemm_wmma<<<Gtok, B256, 0, stream>>>(x_cur16, DIM, wqh + wo, DIM, p_bq + vo,
                                             nullptr, qb, nullptr, DIM, NTOK, DIM, DIM, 0);
        gemm_wmma<<<Gtok, B256, 0, stream>>>(x_cur16, DIM, wksh + wo, DIM, p_bks + vo,
                                             nullptr, ktokb, nullptr, DIM, NTOK, DIM, DIM, 0);
        gemm_wmma<<<Gtok, B256, 0, stream>>>(x_cur16, DIM, wvsh + wo, DIM, p_bvs + vo,
                                             nullptr, vtokb, nullptr, DIM, NTOK, DIM, DIM, 0);
        gemm_wmma<<<Gad, B256, 0, stream>>>(adp16, DIM, wkah + wo, DIM, p_bka + vo,
                                            nullptr, nullptr, kadb, DIM, NAD, DIM, DIM, 0);
        gemm_wmma<<<Gad, B256, 0, stream>>>(adp16, DIM, wvah + wo, DIM, p_bva + vo,
                                            nullptr, nullptr, vadb, DIM, NAD, DIM, DIM, 0);
        gemm_wmma<<<Gta, B256, 0, stream>>>(hts16, DIM, wkth + wo, DIM, p_bkt + vo,
                                            nullptr, nullptr, ktab, DIM, NTA, DIM, DIM, 0);
        gemm_wmma<<<Gta, B256, 0, stream>>>(hts16, DIM, wvth + wo, DIM, p_bvt + vo,
                                            nullptr, nullptr, vtab, DIM, NTA, DIM, DIM, 0);

        rope_kernel<float><<<(NTOK * 448 + 255) / 256, B256, 0, stream>>>(qb, NTOK, TSEQ);
        rope_kernel<float><<<(NTOK * 448 + 255) / 256, B256, 0, stream>>>(ktokb, NTOK, TSEQ);
        rope_kernel<_Float16><<<(NAD * 448 + 255) / 256, B256, 0, stream>>>(kadb, NAD, KADP);
        rope_kernel<_Float16><<<(NTA * 448 + 255) / 256, B256, 0, stream>>>(ktab, NTA, KT);

        attn_kernel<<<BSZ * HEADS, B256, 0, stream>>>(qb, ktokb, vtokb, kadb, vadb,
                                                      ktab, vtab, p_g + blk, attn16);

        // h = Wo @ attn + bo + x ; x = relu(Wf @ LN(h) + bf)
        gemm_wmma<<<Gtok, B256, 0, stream>>>(attn16, DIM, woh + wo, DIM, p_bo + vo,
                                             x_cur, hb, nullptr, DIM, NTOK, DIM, DIM, GF_RES);
        ln_kernel<<<64, B256, 0, stream>>>(hb, lnw + vo, lnb + vo, nullptr, hln16, NTOK, DIM);
        gemm_wmma<<<Gtok, B256, 0, stream>>>(hln16, DIM, wfh + wo, DIM, p_bf + vo,
                                             nullptr, x_cur, x_cur16, DIM, NTOK, DIM, DIM, GF_RELU);
    }

    ln_kernel<<<64, B256, 0, stream>>>(x_cur, ln2w, ln2b, xc32, nullptr, NTOK, DIM);
    final_proj<<<448, B256, 0, stream>>>(xc32, W2, b2, (float*)d_out);
}
